// SparseLinear_34394098106964
// MI455X (gfx1250) — compile-verified, matrix operand order checked by reading, not run on hardware
//
#include <hip/hip_runtime.h>

// SparseLinear y = x @ W^T + b, W = COO(4096x4096, 500k nnz), batch 512.
//
// MI455X strategy: densify W into f16 in L2-resident workspace (32 MB << 192MB L2),
// then run a dense f16 WMMA GEMM with fp32 accumulation. Dense inflates FLOPs 32x
// (0.5 -> 17.2 GFLOP) but WMMA f16 throughput >> VALU fp32, and the whole working
// set (W 32MB + x 4MB + out 8MB) lives in L2, so this beats the L2-gather SpMM
// (~1 GB of gathered traffic) on this chip.

#define IN_F   4096
#define OUT_F  4096
#define NNZ_N  500000
#define BATCH  512

typedef __attribute__((ext_vector_type(16))) _Float16 v16h;
typedef __attribute__((ext_vector_type(8)))  _Float16 v8h;
typedef __attribute__((ext_vector_type(8)))  float    v8f;
typedef __attribute__((ext_vector_type(4)))  unsigned int v4u;

// ---------------------------------------------------------------------------
// Kernel 1: zero the dense f16 weight matrix (32 MB) with b128 stores.
// ---------------------------------------------------------------------------
__global__ void k_zero_w(v4u* __restrict__ w_vec, int n_vec) {
    int i = blockIdx.x * blockDim.x + threadIdx.x;
    if (i < n_vec) {
        v4u z = {0u, 0u, 0u, 0u};
        w_vec[i] = z;
    }
}

// ---------------------------------------------------------------------------
// Kernel 2: convert x fp32 -> f16.
// ---------------------------------------------------------------------------
__global__ void k_convert_x(const float* __restrict__ x, _Float16* __restrict__ xh, int n) {
    int i = blockIdx.x * blockDim.x + threadIdx.x;
    if (i < n) {
        xh[i] = (_Float16)x[i];
    }
}

// ---------------------------------------------------------------------------
// Kernel 3: scatter COO values into dense f16 W with packed-f16 global atomics.
// GLOBAL_ATOMIC_PK_ADD_F16 adds {lo,hi} f16 pair; the untouched half adds +0.0
// which is exact. Duplicate coordinates sum, matching segment_sum semantics.
// ---------------------------------------------------------------------------
__global__ void k_scatter(const long long* __restrict__ idx,   // [2*NNZ]: rows then cols
                          const float* __restrict__ vals,      // [NNZ]
                          _Float16* __restrict__ wh) {
    int i = blockIdx.x * blockDim.x + threadIdx.x;
    if (i < NNZ_N) {
        long long r = idx[i];
        long long c = idx[NNZ_N + i];
        union { _Float16 h; unsigned short u; } cv;
        cv.h = (_Float16)vals[i];
        unsigned int data = (c & 1) ? ((unsigned int)cv.u << 16) : (unsigned int)cv.u;
        const _Float16* p = wh + (size_t)r * IN_F + (size_t)(c & ~1LL);
        asm volatile("global_atomic_pk_add_f16 %0, %1, off"
                     :
                     : "v"(p), "v"(data)
                     : "memory");
    }
}

// ---------------------------------------------------------------------------
// Kernel 4: dense GEMM out[b][r] = sum_c xh[b][c] * wh[r][c] + bias[r]
// via V_WMMA_F32_16X16X32_F16.
//
// 16-bit A-matrix 16x32 lane layout (ISA 7.12.2): lane L<16 -> M=L, K=0..7 &
// 16..23; lane L>=16 -> M=L-16, K=8..15 & 24..31. Both A (x, row-major over c)
// and B (W^T: B[k][n] = W[r0+n][k], i.e. W rows, row-major over c) load as two
// contiguous 16-byte chunks per lane. C/D layout: VGPR i holds M=i (lanes 0-15)
// and M=i+8 (lanes 16-31), N = lane&15.
//
// Block = 256 threads = 8 waves in a 2(M) x 4(N) arrangement; each wave owns a
// 32x64 tile as a 2x4 grid of 16x16 WMMAs. Grid = (4096/256, 512/64) = (16, 8).
// ---------------------------------------------------------------------------
__device__ inline v16h load_frag16(const _Float16* __restrict__ lane_base, int k0) {
    const v8h* p = (const v8h*)(lane_base + k0);
    v8h lo = p[0];   // K offsets 0..7   (8..15 for upper half-lanes)
    v8h hi = p[2];   // K offsets 16..23 (24..31 for upper half-lanes)
    return __builtin_shufflevector(lo, hi, 0, 1, 2, 3, 4, 5, 6, 7,
                                           8, 9, 10, 11, 12, 13, 14, 15);
}

__global__ __launch_bounds__(256) void k_gemm(const _Float16* __restrict__ xh,
                                              const _Float16* __restrict__ wh,
                                              const float* __restrict__ bias,
                                              float* __restrict__ out) {
    const int lane   = threadIdx.x & 31;
    const int wave   = threadIdx.x >> 5;
    const int lane16 = lane & 15;
    const int halfk  = (lane >> 4) & 1;     // 0: K 0..7/16..23, 1: K 8..15/24..31
    const int waveM  = wave & 1;            // 2 waves along batch
    const int waveN  = wave >> 1;           // 4 waves along out-features

    const int b0 = blockIdx.y * 64 + waveM * 32;    // batch base of this wave
    const int r0 = blockIdx.x * 256 + waveN * 64;   // out-row base of this wave

    const _Float16* baseA[2];
    const _Float16* baseB[4];
#pragma unroll
    for (int m = 0; m < 2; ++m)
        baseA[m] = xh + (size_t)(b0 + m * 16 + lane16) * IN_F + halfk * 8;
#pragma unroll
    for (int n = 0; n < 4; ++n)
        baseB[n] = wh + (size_t)(r0 + n * 16 + lane16) * IN_F + halfk * 8;

    v8f c[2][4];
#pragma unroll
    for (int m = 0; m < 2; ++m)
#pragma unroll
        for (int n = 0; n < 4; ++n)
            c[m][n] = (v8f){0.f, 0.f, 0.f, 0.f, 0.f, 0.f, 0.f, 0.f};

#pragma unroll 2
    for (int k0 = 0; k0 < IN_F; k0 += 32) {
        v16h a[2], b[4];
#pragma unroll
        for (int m = 0; m < 2; ++m) a[m] = load_frag16(baseA[m], k0);
#pragma unroll
        for (int n = 0; n < 4; ++n) b[n] = load_frag16(baseB[n], k0);
#pragma unroll
        for (int m = 0; m < 2; ++m)
#pragma unroll
            for (int n = 0; n < 4; ++n)
                c[m][n] = __builtin_amdgcn_wmma_f32_16x16x32_f16(
                    /*neg_a=*/false, a[m], /*neg_b=*/false, b[n],
                    /*c_mod=*/(short)0, c[m][n],
                    /*reuse_a=*/false, /*reuse_b=*/false);
    }

    // Epilogue: fused bias add. D VGPR i: M = i + 8*halfk, N = lane16.
#pragma unroll
    for (int n = 0; n < 4; ++n) {
        const int col = r0 + n * 16 + lane16;
        const float bv = bias[col];
#pragma unroll
        for (int m = 0; m < 2; ++m) {
#pragma unroll
            for (int i = 0; i < 8; ++i) {
                const int row = b0 + m * 16 + halfk * 8 + i;
                out[(size_t)row * OUT_F + col] = c[m][n][i] + bv;
            }
        }
    }
}

// ---------------------------------------------------------------------------
// Launch. ws layout: [0,32MB) dense f16 W ; [32MB,36MB) f16 x.
// ---------------------------------------------------------------------------
extern "C" void kernel_launch(void* const* d_in, const int* in_sizes, int n_in,
                              void* d_out, int out_size, void* d_ws, size_t ws_size,
                              hipStream_t stream) {
    const float*     x    = (const float*)d_in[0];      // [512, 4096]
    const float*     wv   = (const float*)d_in[1];      // [500000]
    const float*     bias = (const float*)d_in[2];      // [4096]
    const long long* idx  = (const long long*)d_in[3];  // [2, 500000] int64
    float*           out  = (float*)d_out;              // [512, 4096]

    _Float16* wh = (_Float16*)d_ws;
    _Float16* xh = (_Float16*)((char*)d_ws + (size_t)OUT_F * IN_F * sizeof(_Float16));

    // 1) zero dense W: 4096*4096 halves = 2,097,152 x 16B vectors
    {
        int n_vec = (OUT_F * IN_F) / 8;
        k_zero_w<<<(n_vec + 255) / 256, 256, 0, stream>>>((v4u*)wh, n_vec);
    }
    // 2) convert x to f16
    {
        int n = BATCH * IN_F;
        k_convert_x<<<(n + 255) / 256, 256, 0, stream>>>(x, xh, n);
    }
    // 3) scatter COO into dense W (packed f16 atomic add)
    k_scatter<<<(NNZ_N + 255) / 256, 256, 0, stream>>>(idx, wv, wh);

    // 4) WMMA GEMM + bias
    dim3 grid(OUT_F / 256, BATCH / 64);
    k_gemm<<<grid, 256, 0, stream>>>(xh, wh, bias, out);
}
